// MambaLayer_1_53824530154110
// MI455X (gfx1250) — matrix operand order repaired and checked by reading, not compiled, not run on hardware
//
#include <hip/hip_runtime.h>
#include <math.h>

typedef __attribute__((ext_vector_type(16))) _Float16 v16h;
typedef __attribute__((ext_vector_type(8)))  _Float16 v8h;
typedef __attribute__((ext_vector_type(8)))  float    v8f;

#define DIRS   4
#define LSEQ   4096
#define DIMC   512
#define DIC    128
#define MROWS  (DIRS * 4 * LSEQ)   // 65536
#define NCHUNK 64
#define LCHUNK 64

__device__ __forceinline__ v8f  zero8()  { v8f z = {};  return z; }
__device__ __forceinline__ v16h zero16() { v16h z = {}; return z; }

// A fragment (16x32 f16, MxK) from an f16 row pointer:
// elements 0..15 = K(half*16 + 0..15)
__device__ __forceinline__ v16h load_a_frag(const _Float16* __restrict__ rowptr, int kbase, int half) {
  v16h a;
  const v8h lo = *(const v8h*)(rowptr + kbase + half * 16);
  const v8h hi = *(const v8h*)(rowptr + kbase + half * 16 + 8);
#pragma unroll
  for (int e = 0; e < 8; ++e) { a[e] = lo[e]; a[8 + e] = hi[e]; }
  return a;
}

// B fragment (32x16 f16, KxN) from row-major f16 weight W[n][k]:
// lane holds col n, elements = 16 contiguous K at (kbase + half*16)  -> one aligned 32B load
__device__ __forceinline__ v16h load_b_frag(const _Float16* __restrict__ w, int ld, int n,
                                            int kbase, int half) {
  return *(const v16h*)(w + (size_t)n * ld + kbase + half * 16);
}

__device__ __forceinline__ v8f wmma_f16(v16h a, v16h b, v8f c) {
  return __builtin_amdgcn_wmma_f32_16x16x32_f16(false, a, false, b, (short)0, c, false, false);
}

// ---------------- 0) fp32 -> f16 convert (weights + x), one-time ----------------
__global__ void k_cvt16(const float* __restrict__ src, _Float16* __restrict__ dst, int n) {
  int i = blockIdx.x * blockDim.x + threadIdx.x;
  if (i < n) dst[i] = (_Float16)src[i];
}

// ---------------- 1) in_proj with direction gather: xz = gather(x16) @ in_w.T ---
__global__ void __launch_bounds__(128) k_in_proj(const _Float16* __restrict__ x16,
                                                 const _Float16* __restrict__ in_w, // (4,256,128) f16
                                                 float* __restrict__ xz) {          // (65536,256) f32
  const int lane = threadIdx.x & 31;
  const int wid  = threadIdx.x >> 5;
  const int half = lane >> 4;
  const int ml   = lane & 15;
  const int mtile = blockIdx.x * 4 + wid;   // 4096 tiles
  const int g0 = mtile * 16;
  const int dir = g0 >> 14;                  // constant within tile

  int g   = g0 + ml;
  int rem = g & 16383;
  int b   = rem >> 12;
  int i   = rem & 4095;
  int srcpos, cb;
  if      (dir == 0) { srcpos = i;                         cb = 0;   }
  else if (dir == 1) { srcpos = LSEQ - 1 - i;              cb = 128; }
  else if (dir == 2) { srcpos = (i & 63) * 64 + (i >> 6);  cb = 256; }
  else { int j = LSEQ - 1 - i; srcpos = (j & 63) * 64 + (j >> 6); cb = 384; }
  const _Float16* rowptr = x16 + ((size_t)b * LSEQ + srcpos) * DIMC + cb;

  v16h afr[4];
#pragma unroll
  for (int ks = 0; ks < 4; ++ks) afr[ks] = load_a_frag(rowptr, ks * 32, half);

  const _Float16* w = in_w + (size_t)dir * 256 * 128;
  for (int grp = 0; grp < 4; ++grp) {          // N=256 -> 4 groups of 64 cols
    v8f acc[4];
#pragma unroll
    for (int j = 0; j < 4; ++j) acc[j] = zero8();
#pragma unroll
    for (int ks = 0; ks < 4; ++ks) {
#pragma unroll
      for (int j = 0; j < 4; ++j) {
        int n = grp * 64 + j * 16 + ml;
        acc[j] = wmma_f16(afr[ks], load_b_frag(w, 128, n, ks * 32, half), acc[j]);
      }
    }
#pragma unroll
    for (int j = 0; j < 4; ++j) {
      int nb = grp * 64 + j * 16;
#pragma unroll
      for (int r = 0; r < 8; ++r)
        xz[(size_t)(g0 + half * 8 + r) * 256 + nb + ml] = acc[j][r];
    }
  }
}

// ---------------- 2) depthwise causal conv(4) + bias + SiLU (f32 + f16 out) -----
__global__ void k_conv_silu(const float* __restrict__ xz, const float* __restrict__ conv_w,
                            const float* __restrict__ conv_b,
                            float* __restrict__ xc32, _Float16* __restrict__ xc16) {
  int idx = blockIdx.x * blockDim.x + threadIdx.x;     // MROWS*128
  if (idx >= MROWS * DIC) return;
  int d = idx & 127;
  int g = idx >> 7;
  int dir = g >> 14;
  int i = g & 4095;
  float s = conv_b[dir * 128 + d];
  const float* w = conv_w + ((size_t)dir * 128 + d) * 4;
#pragma unroll
  for (int t = 0; t < 4; ++t) {
    int src = i - 3 + t;
    if (src >= 0) s += w[t] * xz[(size_t)(g - 3 + t) * 256 + d];
  }
  float v = s / (1.0f + __expf(-s));                   // SiLU
  xc32[(size_t)g * 128 + d] = v;
  xc16[(size_t)g * 128 + d] = (_Float16)v;
}

// ---------------- 3) x_proj: xdb = xc16 @ xp_w.T  (N=40, padded to 48) ----------
__global__ void __launch_bounds__(128) k_x_proj(const _Float16* __restrict__ xc16,
                                                const _Float16* __restrict__ xp_w, // (4,40,128) f16
                                                float* __restrict__ xdb) {         // (65536,40)
  const int lane = threadIdx.x & 31;
  const int wid  = threadIdx.x >> 5;
  const int half = lane >> 4;
  const int ml   = lane & 15;
  const int mtile = blockIdx.x * 4 + wid;
  const int g0 = mtile * 16;
  const int dir = g0 >> 14;
  const _Float16* rowptr = xc16 + (size_t)(g0 + ml) * 128;

  v16h afr[4];
#pragma unroll
  for (int ks = 0; ks < 4; ++ks) afr[ks] = load_a_frag(rowptr, ks * 32, half);

  const _Float16* w = xp_w + (size_t)dir * 40 * 128;
  v8f acc[3];
#pragma unroll
  for (int j = 0; j < 3; ++j) acc[j] = zero8();
#pragma unroll
  for (int ks = 0; ks < 4; ++ks) {
#pragma unroll
    for (int j = 0; j < 3; ++j) {
      int n = j * 16 + ml;
      v16h bfr = (n < 40) ? load_b_frag(w, 128, n, ks * 32, half) : zero16();
      acc[j] = wmma_f16(afr[ks], bfr, acc[j]);
    }
  }
#pragma unroll
  for (int j = 0; j < 3; ++j) {
    int n = j * 16 + ml;
    if (n < 40) {
#pragma unroll
      for (int r = 0; r < 8; ++r)
        xdb[(size_t)(g0 + half * 8 + r) * 40 + n] = acc[j][r];
    }
  }
}

// ---------------- 4) dt = softplus(xdb[:,:8] @ dt_w.T + dt_b) -------------------
__global__ void k_dt_proj(const float* __restrict__ xdb, const float* __restrict__ dt_w,
                          const float* __restrict__ dt_b, float* __restrict__ dt) {
  int idx = blockIdx.x * blockDim.x + threadIdx.x;
  if (idx >= MROWS * DIC) return;
  int d = idx & 127;
  int g = idx >> 7;
  int dir = g >> 14;
  const float* row = xdb + (size_t)g * 40;
  const float* w = dt_w + ((size_t)dir * 128 + d) * 8;
  float s = dt_b[dir * 128 + d];
#pragma unroll
  for (int k = 0; k < 8; ++k) s += row[k] * w[k];
  dt[(size_t)g * 128 + d] = (s > 20.f) ? s : log1pf(__expf(s));
}

// ---------------- 5) scan pass 1: per-chunk aggregates (prod a, u) --------------
__global__ void __launch_bounds__(128) k_scan_reduce(const float* __restrict__ dt,
    const float* __restrict__ xc, const float* __restrict__ xdb,
    const float* __restrict__ A_log, float* __restrict__ Aagg, float* __restrict__ Uagg) {
  const int seq   = blockIdx.x >> 6;       // dir*4+b
  const int chunk = blockIdx.x & 63;
  const int dir   = seq >> 2;
  const int d     = threadIdx.x;
  float a_d[16], ap[16], u[16];
#pragma unroll
  for (int n = 0; n < 16; ++n) {
    a_d[n] = -__expf(A_log[((size_t)dir * 128 + d) * 16 + n]);
    ap[n] = 1.f; u[n] = 0.f;
  }
  const int gbase = seq * LSEQ + chunk * LCHUNK;
  for (int t = 0; t < LCHUNK; ++t) {
    int g = gbase + t;
    if (t + 8 < LCHUNK) {                                  // keep stream ahead of exp chain
      __builtin_prefetch(dt + (size_t)(g + 8) * 128 + d, 0, 1);
      __builtin_prefetch(xdb + (size_t)(g + 8) * 40, 0, 1);
    }
    float dtv = dt[(size_t)g * 128 + d];
    float xv  = xc[(size_t)g * 128 + d];
    float du  = dtv * xv;
    const float* Bp = xdb + (size_t)g * 40 + 8;
#pragma unroll
    for (int n = 0; n < 16; ++n) {
      float coef = __expf(dtv * a_d[n]);
      u[n]  = coef * u[n] + du * Bp[n];
      ap[n] *= coef;
    }
  }
  size_t ob = (size_t)blockIdx.x * 2048 + d * 16;
#pragma unroll
  for (int n = 0; n < 16; ++n) { Aagg[ob + n] = ap[n]; Uagg[ob + n] = u[n]; }
}

// ---------------- 6) scan pass 2: serial prefix over 64 chunk aggregates --------
__global__ void k_scan_prefix(const float* __restrict__ Aagg, const float* __restrict__ Uagg,
                              float* __restrict__ Hinit) {
  int idx = blockIdx.x * blockDim.x + threadIdx.x;   // 16*2048 = 32768
  if (idx >= 16 * 2048) return;
  int seq = idx >> 11;
  int s   = idx & 2047;
  float h = 0.f;
  for (int c = 0; c < NCHUNK; ++c) {
    size_t p = (size_t)(seq * NCHUNK + c) * 2048 + s;
    Hinit[p] = h;
    h = Aagg[p] * h + Uagg[p];
  }
}

// ---------------- 7) scan pass 3: apply + y = h·C, fuse (+xc*D)*silu(z) ---------
__global__ void __launch_bounds__(128) k_scan_apply(const float* __restrict__ dt,
    const float* __restrict__ xc, const float* __restrict__ xdb, const float* __restrict__ xz,
    const float* __restrict__ A_log, const float* __restrict__ Dp,
    const float* __restrict__ Hinit, _Float16* __restrict__ ypre16) {
  const int seq   = blockIdx.x >> 6;
  const int chunk = blockIdx.x & 63;
  const int dir   = seq >> 2;
  const int d     = threadIdx.x;
  float a_d[16], h[16];
  size_t hb = (size_t)blockIdx.x * 2048 + d * 16;
#pragma unroll
  for (int n = 0; n < 16; ++n) {
    a_d[n] = -__expf(A_log[((size_t)dir * 128 + d) * 16 + n]);
    h[n] = Hinit[hb + n];
  }
  const float Dv = Dp[dir * 128 + d];
  const int gbase = seq * LSEQ + chunk * LCHUNK;
  for (int t = 0; t < LCHUNK; ++t) {
    int g = gbase + t;
    if (t + 8 < LCHUNK) {
      __builtin_prefetch(dt + (size_t)(g + 8) * 128 + d, 0, 1);
      __builtin_prefetch(xdb + (size_t)(g + 8) * 40, 0, 1);
      __builtin_prefetch(xz + (size_t)(g + 8) * 256 + 128 + d, 0, 1);
    }
    float dtv = dt[(size_t)g * 128 + d];
    float xv  = xc[(size_t)g * 128 + d];
    float du  = dtv * xv;
    const float* Bp = xdb + (size_t)g * 40 + 8;
    const float* Cp = xdb + (size_t)g * 40 + 24;
    float y = 0.f;
#pragma unroll
    for (int n = 0; n < 16; ++n) {
      float coef = __expf(dtv * a_d[n]);
      h[n] = coef * h[n] + du * Bp[n];
      y += h[n] * Cp[n];
    }
    float yv = y + xv * Dv;
    float zv = xz[(size_t)g * 256 + 128 + d];
    ypre16[(size_t)g * 128 + d] = (_Float16)(yv * (zv / (1.f + __expf(-zv))));
  }
}

// ---------------- 8) out_proj GEMM + un-permute scatter into y512 (f16) ---------
__global__ void __launch_bounds__(128) k_out_proj(const _Float16* __restrict__ ypre16,
    const _Float16* __restrict__ out_w,  // (4,128,128) f16
    _Float16* __restrict__ y512) {       // [b][pos][512] f16
  const int lane = threadIdx.x & 31;
  const int wid  = threadIdx.x >> 5;
  const int half = lane >> 4;
  const int ml   = lane & 15;
  const int mtile = blockIdx.x * 4 + wid;
  const int g0 = mtile * 16;
  const int dir = g0 >> 14;
  const _Float16* rowptr = ypre16 + (size_t)(g0 + ml) * 128;

  v16h afr[4];
#pragma unroll
  for (int ks = 0; ks < 4; ++ks) afr[ks] = load_a_frag(rowptr, ks * 32, half);

  const _Float16* w = out_w + (size_t)dir * 128 * 128;
  for (int grp = 0; grp < 2; ++grp) {
    v8f acc[4];
#pragma unroll
    for (int j = 0; j < 4; ++j) acc[j] = zero8();
#pragma unroll
    for (int ks = 0; ks < 4; ++ks) {
#pragma unroll
      for (int j = 0; j < 4; ++j) {
        int n = grp * 64 + j * 16 + ml;
        acc[j] = wmma_f16(afr[ks], load_b_frag(w, 128, n, ks * 32, half), acc[j]);
      }
    }
#pragma unroll
    for (int j = 0; j < 4; ++j) {
      int n = grp * 64 + j * 16 + ml;
#pragma unroll
      for (int r = 0; r < 8; ++r) {
        int m = g0 + half * 8 + r;
        int rem = m & 16383;
        int b = rem >> 12;
        int i = rem & 4095;
        int p;
        if      (dir == 0) p = i;
        else if (dir == 1) p = 4095 - i;
        else if (dir == 2) p = (i & 63) * 64 + (i >> 6);
        else { int q = 4095 - i; p = (q & 63) * 64 + (q >> 6); }
        y512[((size_t)b * LSEQ + p) * DIMC + dir * 128 + n] = (_Float16)acc[j][r];
      }
    }
  }
}

// ---------------- 9) final: out = clip(y512 @ proj_w.T + b), nan->0 -------------
__global__ void __launch_bounds__(128) k_final(const _Float16* __restrict__ y512,
    const _Float16* __restrict__ pw, const float* __restrict__ pb, float* __restrict__ out) {
  const int lane = threadIdx.x & 31;
  const int wid  = threadIdx.x >> 5;
  const int half = lane >> 4;
  const int ml   = lane & 15;
  const int mtile = blockIdx.x * 4 + wid;     // 1024 tiles (M=16384)
  const int g0 = mtile * 16;
  const _Float16* rowptr = y512 + (size_t)(g0 + ml) * DIMC;

  for (int grp = 0; grp < 8; ++grp) {          // N=512 -> 8 groups of 64
    v8f acc[4];
#pragma unroll
    for (int j = 0; j < 4; ++j) acc[j] = zero8();
    for (int ks = 0; ks < 16; ++ks) {          // K=512
      v16h afr = load_a_frag(rowptr, ks * 32, half);
#pragma unroll
      for (int j = 0; j < 4; ++j) {
        int n = grp * 64 + j * 16 + ml;
        acc[j] = wmma_f16(afr, load_b_frag(pw, 512, n, ks * 32, half), acc[j]);
      }
    }
#pragma unroll
    for (int j = 0; j < 4; ++j) {
      int n = grp * 64 + j * 16 + ml;
#pragma unroll
      for (int r = 0; r < 8; ++r) {
        float v = acc[j][r] + pb[n];
        if (v != v) v = 0.f;                               // nan -> 0
        v = fminf(fmaxf(v, -1000.f), 1000.f);              // clip (also maps +-inf)
        out[(size_t)(g0 + half * 8 + r) * DIMC + n] = v;
      }
    }
  }
}

extern "C" void kernel_launch(void* const* d_in, const int* in_sizes, int n_in,
                              void* d_out, int out_size, void* d_ws, size_t ws_size,
                              hipStream_t stream) {
  const float* x      = (const float*)d_in[0];
  const float* in_w   = (const float*)d_in[1];
  const float* conv_w = (const float*)d_in[2];
  const float* conv_b = (const float*)d_in[3];
  const float* xp_w   = (const float*)d_in[4];
  const float* dt_w   = (const float*)d_in[5];
  const float* dt_b   = (const float*)d_in[6];
  const float* A_log  = (const float*)d_in[7];
  const float* Dp     = (const float*)d_in[8];
  const float* out_w  = (const float*)d_in[9];
  const float* pw     = (const float*)d_in[10];
  const float* pb     = (const float*)d_in[11];
  float* out = (float*)d_out;

  // ---- workspace layout (256B-aligned sub-buffers) ----
  char* base = (char*)d_ws;
  size_t off = 0;
  auto take = [&](size_t bytes) -> void* {
    void* p = base + off;
    off += (bytes + 255) & ~(size_t)255;
    return p;
  };
  float*    xz     = (float*)   take((size_t)16777216 * 4);   // (65536,256) f32
  float*    xc32   = (float*)   take((size_t)8388608  * 4);   // (65536,128) f32
  float*    xdb    = (float*)   take((size_t)2621440  * 4);   // (65536,40)  f32
  float*    dt     = (float*)   take((size_t)8388608  * 4);   // (65536,128) f32
  float*    Aagg   = (float*)   take((size_t)2097152  * 4);
  float*    Uagg   = (float*)   take((size_t)2097152  * 4);
  float*    Hinit  = (float*)   take((size_t)2097152  * 4);
  _Float16* x16    = (_Float16*)take((size_t)8388608  * 2);   // x in f16
  _Float16* xc16   = (_Float16*)take((size_t)8388608  * 2);
  _Float16* ypre16 = (_Float16*)take((size_t)8388608  * 2);
  _Float16* y51216 = (_Float16*)take((size_t)8388608  * 2);
  _Float16* in_w16 = (_Float16*)take((size_t)131072   * 2);
  _Float16* xp_w16 = (_Float16*)take((size_t)20480    * 2);
  _Float16* out_w16= (_Float16*)take((size_t)65536    * 2);
  _Float16* pw16   = (_Float16*)take((size_t)262144   * 2);

  // one-time f16 conversions (x + all GEMM weights)
  k_cvt16<<<32768, 256, 0, stream>>>(x,     x16,     8388608);
  k_cvt16<<<512,   256, 0, stream>>>(in_w,  in_w16,  131072);
  k_cvt16<<<80,    256, 0, stream>>>(xp_w,  xp_w16,  20480);
  k_cvt16<<<256,   256, 0, stream>>>(out_w, out_w16, 65536);
  k_cvt16<<<1024,  256, 0, stream>>>(pw,    pw16,    262144);

  k_in_proj    <<<1024,  128, 0, stream>>>(x16, in_w16, xz);
  k_conv_silu  <<<32768, 256, 0, stream>>>(xz, conv_w, conv_b, xc32, xc16);
  k_x_proj     <<<1024,  128, 0, stream>>>(xc16, xp_w16, xdb);
  k_dt_proj    <<<32768, 256, 0, stream>>>(xdb, dt_w, dt_b, dt);
  k_scan_reduce<<<1024,  128, 0, stream>>>(dt, xc32, xdb, A_log, Aagg, Uagg);
  k_scan_prefix<<<128,   256, 0, stream>>>(Aagg, Uagg, Hinit);
  k_scan_apply <<<1024,  128, 0, stream>>>(dt, xc32, xdb, xz, A_log, Dp, Hinit, ypre16);
  k_out_proj   <<<1024,  128, 0, stream>>>(ypre16, out_w16, y51216);
  k_final      <<<256,   128, 0, stream>>>(y51216, pw16, pb, out);
}
